// EdgeLayer_79894981640517
// MI455X (gfx1250) — matrix operand order repaired and checked by reading, not compiled, not run on hardware
//
#include <hip/hip_runtime.h>

typedef __bf16 bf16;
typedef bf16 bf16x16 __attribute__((ext_vector_type(16)));
typedef float f32x8 __attribute__((ext_vector_type(8)));
typedef float f32x4 __attribute__((ext_vector_type(4)));
typedef unsigned u32x4 __attribute__((ext_vector_type(4)));

#define EN 600000
#define SI 128
#define VIN 16
#define SE 32
#define VE 4
#define EPSF 1e-8f
#define LNEPS 1e-5f

#define EPW 16              // edges per wave
#define WAVES 8             // waves per block
#define EPB (EPW*WAVES)     // 128 edges per block
#define SS_STRIDE 352       // bf16 elems, covers m0's 324-wide input
#define SV_STRIDE 64
#define SOUT_ELEMS (48*48)  // f32 staging scratch

// per-wave private LDS slab: no inter-wave sharing -> no barriers needed
struct WaveMem {
  bf16  sS[EPW * SS_STRIDE];     // 11264 B  scalar-channel A rows (bf16)
  bf16  sV[EPW * 3 * SV_STRIDE]; //  6144 B  vector-channel A rows (bf16), (edge,t) major
  float sOut[SOUT_ELEMS];        //  9216 B  f32 WMMA staging
  float sSf[EPW * SE];           //  2048 B  x_s f32 (kept for 2nd residual)
  float sVf[EPW * VE * 3];       //   768 B  x_v f32 (kept for 2nd residual)
};                               // 29440 B per wave; x8 = 235520 B < 320KB

#define WSYNC() asm volatile("" ::: "memory")

// ---- bf16 helpers -------------------------------------------------------
__device__ __forceinline__ unsigned pack2(float a, float b) {
  union { bf16 h[2]; unsigned u; } z;
  z.h[0] = (bf16)a; z.h[1] = (bf16)b;
  return z.u;
}
__device__ __forceinline__ unsigned short f2bfbits(float x) {
  union { bf16 h; unsigned short s; } z; z.h = (bf16)x; return z.s;
}

// ---- WMMA fragment loaders (16x16x32 bf16, wave32 layouts) --------------
// A (16x32): lane l -> row l&15; K runs [ (l>>4)*8 .. +8 ) and [16+(l>>4)*8 .. +8)
__device__ __forceinline__ bf16x16 load_a(const bf16* tile, int stride, int kc, int lane) {
  int m = lane & 15, h8 = (lane >> 4) << 3;
  const bf16* p = tile + m * stride + kc + h8;
  union { u32x4 q[2]; bf16x16 f; } u;
  u.q[0] = *(const u32x4*)(p);        // ds_load_b128
  u.q[1] = *(const u32x4*)(p + 16);   // ds_load_b128
  return u.f;
}
// B (32x16): W stored N x Kpad row-major (== jax weight row-major).
// lane l -> col l&15; K runs [ kc+(l>>4)*8 .. +8 ) and [ kc+16+(l>>4)*8 .. +8 )
__device__ __forceinline__ bf16x16 load_b(const bf16* W, int kpad, int nt, int kc, int lane) {
  const bf16* p = W + (size_t)(nt * 16 + (lane & 15)) * kpad + kc + ((lane >> 4) << 3);
  union { u32x4 q[2]; bf16x16 f; } u;
  u.q[0] = *(const u32x4*)(p);        // global_load_b128
  u.q[1] = *(const u32x4*)(p + 16);   // global_load_b128
  return u.f;
}

// C = A*B (+bias, optional relu), staged as f32 into LDS scratch.
// C layout: VGPR r of lane l -> row mt*16 + (l>>4)*8 + r, col nt*16 + (l&15)
// NOTE: bias/relu guards fold away at compile time after inlining with
// literal nullptr/false args -> no "+0" VALU pass on the no-bias matmuls.
__device__ __forceinline__ void mm_stage(const bf16* A, int strideA, int mtiles,
                                         const bf16* W, int kchunks, int ntiles,
                                         float* out, int ostride,
                                         const float* bias, bool relu, int lane) {
  int kpad = kchunks * 32;
  for (int mt = 0; mt < mtiles; ++mt) {
    const bf16* At = A + mt * 16 * strideA;
    for (int nt = 0; nt < ntiles; ++nt) {
      f32x8 acc = {0.f,0.f,0.f,0.f,0.f,0.f,0.f,0.f};
      for (int kc = 0; kc < kchunks; ++kc) {
        bf16x16 a = load_a(At, strideA, kc * 32, lane);
        bf16x16 b = load_b(W, kpad, nt, kc * 32, lane);
        acc = __builtin_amdgcn_wmma_f32_16x16x32_bf16(false, a, false, b,
                                                      (short)0, acc, false, false);
      }
      int col = nt * 16 + (lane & 15);
      int rb = mt * 16 + ((lane >> 4) << 3);
      if (bias) {
        float bv = bias[col];
#pragma unroll
        for (int r = 0; r < 8; ++r) {
          float d = acc[r] + bv;
          if (relu) d = fmaxf(d, 0.0f);
          out[(rb + r) * ostride + col] = d;
        }
      } else {
#pragma unroll
        for (int r = 0; r < 8; ++r) {
          float d = acc[r];
          if (relu) d = fmaxf(d, 0.0f);
          out[(rb + r) * ostride + col] = d;
        }
      }
    }
  }
}

// vn = ||vh|| over t (3 rows) -> sS[.. si+h]; vh bf16 -> sV cols 0..H-1
__device__ __forceinline__ void vn_phase(WaveMem* wm, int row, int half, int H, int si) {
  int hh = H >> 1;
  for (int q = 0; q < hh; ++q) {
    int h = half * hh + q;
    float a0 = wm->sOut[(row*3+0)*48 + h];
    float a1 = wm->sOut[(row*3+1)*48 + h];
    float a2 = wm->sOut[(row*3+2)*48 + h];
    float nn = fmaxf(a0*a0 + a1*a1 + a2*a2, EPSF);
    wm->sS[row*SS_STRIDE + si + h] = (bf16)sqrtf(nn);
    wm->sV[(row*3+0)*SV_STRIDE + h] = (bf16)a0;
    wm->sV[(row*3+1)*SV_STRIDE + h] = (bf16)a1;
    wm->sV[(row*3+2)*SV_STRIDE + h] = (bf16)a2;
  }
}

// v_out * sigmoid(||v_out||) -> sV cols 0..VO-1
__device__ __forceinline__ void gate_phase(WaveMem* wm, int row, int half, int VO) {
  int vv = VO >> 1;
  for (int q = 0; q < vv; ++q) {
    int o = half * vv + q;
    float x = wm->sOut[(row*3+0)*16 + o];
    float y = wm->sOut[(row*3+1)*16 + o];
    float z = wm->sOut[(row*3+2)*16 + o];
    float nrm = sqrtf(fmaxf(x*x + y*y + z*z, EPSF));
    float g = 1.0f / (1.0f + __expf(-nrm));
    wm->sV[(row*3+0)*SV_STRIDE + o] = (bf16)(x*g);
    wm->sV[(row*3+1)*SV_STRIDE + o] = (bf16)(y*g);
    wm->sV[(row*3+2)*SV_STRIDE + o] = (bf16)(z*g);
  }
}

// staged s f32 -> bf16 into sS cols [0,SO); zero cols [SO, zlim)
__device__ __forceinline__ void s_writeback(WaveMem* wm, int row, int half, int SO, int zlim) {
  int c0 = half * (SO >> 1);
  for (int q = 0; q < (SO >> 1); q += 2) {
    int c = c0 + q;
    *(unsigned*)(wm->sS + row*SS_STRIDE + c) =
        pack2(wm->sOut[row*SO + c], wm->sOut[row*SO + c + 1]);
  }
  int zh = (zlim - SO) >> 1;                 // per-half zero count (even in all uses)
  int z0 = SO + half * zh;
  for (int q = 0; q < zh; q += 2)
    *(unsigned*)(wm->sS + row*SS_STRIDE + z0 + q) = 0u;
}

// ---- weight workspace offsets (bf16 elems) ------------------------------
#define OFF_WH0 0       /* 48x64   */
#define OFF_WS0 3072    /* 32x352  */
#define OFF_WV0 14336   /* 16x64   */
#define OFF_WH1 15360   /* 16x32   */
#define OFF_WS1 15872   /* 32x64   */
#define OFF_WV1 17920   /* 16x32   */
#define OFF_WH2 18432
#define OFF_WS2 18944
#define OFF_WV2 20992
#define OFF_WH3 21504   /* 16x32   */
#define OFF_WS3 22016   /* 128x64  */
#define OFF_WV3 30208   /* 16x32   */
#define OFF_WH4 30720   /* 16x32   */
#define OFF_WS4 31232   /* 32x160  */
#define OFF_WV4 36352   /* 16x32   */

struct KArgs {
  const float *hVs, *hVv, *hEs, *hEv;
  const int *eidx;
  const float *bs[5];
  const float *g0, *b0, *g1, *b1;
  const unsigned short *W;   // bf16 bits
  float *outS, *outV;
};

__global__ void prep_weights(const float* src, unsigned short* dst,
                             int N, int K, int Npad, int Kpad) {
  int idx = blockIdx.x * blockDim.x + threadIdx.x;
  if (idx >= Npad * Kpad) return;
  int n = idx / Kpad, k = idx - n * Kpad;
  float v = (n < N && k < K) ? src[n * K + k] : 0.0f;
  dst[idx] = f2bfbits(v);
}

__global__ __launch_bounds__(256, 1) void edge_gvp_kernel(KArgs A) {
  extern __shared__ char g_lds[];
  int tid = threadIdx.x;
  int wave = tid >> 5, lane = tid & 31;
  int row = lane & 15, half = lane >> 4;
  WaveMem* wm = (WaveMem*)(g_lds + (size_t)wave * sizeof(WaveMem));
  const bf16* W = (const bf16*)A.W;

  long e = (long)blockIdx.x * EPB + wave * EPW + row;
  bool valid = e < EN;
  int i0 = 0, j0 = 0;
  if (valid) { i0 = A.eidx[e]; j0 = A.eidx[EN + e]; }

  // ---------------- gather: build bf16 A rows in LDS ----------------
  {
    const float* vsi = A.hVs + (long)i0 * SI;
    const float* vsj = A.hVs + (long)j0 * SI;
    const float* ves = A.hEs + e * SE;
    bf16* srow = wm->sS + row * SS_STRIDE;
    for (int q = 0; q < 144; q += 2) {              // halves split [0,288)
      int c = half * 144 + q;
      float x0 = 0.f, x1 = 0.f;
      if (valid) {
        x0 = (c < 128) ? vsi[c] : (c < 160) ? ves[c - 128] : vsj[c - 160];
        int c1 = c + 1;
        x1 = (c1 < 128) ? vsi[c1] : (c1 < 160) ? ves[c1 - 128] : vsj[c1 - 160];
      }
      *(unsigned*)(srow + c) = pack2(x0, x1);
    }
    for (int q = 0; q < 32; q += 2)                 // zero [288,352)
      *(unsigned*)(srow + 288 + half * 32 + q) = 0u;

    const float* vvi = A.hVv + (long)i0 * (VIN * 3);
    const float* vvj = A.hVv + (long)j0 * (VIN * 3);
    const float* vev = A.hEv + e * (VE * 3);
    for (int q = 0; q < 18; ++q) {                  // 36 vectors, halves split
      int vv = half * 18 + q;
      float x = 0.f, y = 0.f, z = 0.f;
      if (valid) {
        const float* s = (vv < 16) ? (vvi + vv * 3)
                       : (vv < 20) ? (vev + (vv - 16) * 3)
                                   : (vvj + (vv - 20) * 3);
        x = s[0]; y = s[1]; z = s[2];
      }
      wm->sV[(row*3+0)*SV_STRIDE + vv] = (bf16)x;
      wm->sV[(row*3+1)*SV_STRIDE + vv] = (bf16)y;
      wm->sV[(row*3+2)*SV_STRIDE + vv] = (bf16)z;
    }
    for (int q = 0; q < 14; ++q) {                  // zero [36,64)
      int c = 36 + half * 14 + q;
      wm->sV[(row*3+0)*SV_STRIDE + c] = (bf16)0.f;
      wm->sV[(row*3+1)*SV_STRIDE + c] = (bf16)0.f;
      wm->sV[(row*3+2)*SV_STRIDE + c] = (bf16)0.f;
    }
  }
  WSYNC();

  // ================= m0: (si=288,vi=36) -> (32,4), act =================
  mm_stage(wm->sV, SV_STRIDE, 3, W + OFF_WH0, 2, 3, wm->sOut, 48, nullptr, false, lane);
  WSYNC();
  vn_phase(wm, row, half, 36, 288);
  WSYNC();
  mm_stage(wm->sS, SS_STRIDE, 1, W + OFF_WS0, 11, 2, wm->sOut, 32, A.bs[0], true, lane);
  WSYNC();
  s_writeback(wm, row, half, 32, 64);
  WSYNC();
  mm_stage(wm->sV, SV_STRIDE, 3, W + OFF_WV0, 2, 1, wm->sOut, 16, nullptr, false, lane);
  WSYNC();
  gate_phase(wm, row, half, 4);
  for (int q = 0; q < 14; ++q) {                    // zero sV [4,32) once
    int c = 4 + half * 14 + q;
    wm->sV[(row*3+0)*SV_STRIDE + c] = (bf16)0.f;
    wm->sV[(row*3+1)*SV_STRIDE + c] = (bf16)0.f;
    wm->sV[(row*3+2)*SV_STRIDE + c] = (bf16)0.f;
  }
  WSYNC();

  // ================= m1: (32,4) -> (32,4), act =================
  mm_stage(wm->sV, SV_STRIDE, 3, W + OFF_WH1, 1, 1, wm->sOut, 48, nullptr, false, lane);
  WSYNC();
  vn_phase(wm, row, half, 4, 32);
  WSYNC();
  mm_stage(wm->sS, SS_STRIDE, 1, W + OFF_WS1, 2, 2, wm->sOut, 32, A.bs[1], true, lane);
  WSYNC();
  s_writeback(wm, row, half, 32, 32);
  WSYNC();
  mm_stage(wm->sV, SV_STRIDE, 3, W + OFF_WV1, 1, 1, wm->sOut, 16, nullptr, false, lane);
  WSYNC();
  gate_phase(wm, row, half, 4);
  WSYNC();

  // ================= m2: (32,4) -> (32,4), no act; residual + LN0 =======
  mm_stage(wm->sV, SV_STRIDE, 3, W + OFF_WH2, 1, 1, wm->sOut, 48, nullptr, false, lane);
  WSYNC();
  vn_phase(wm, row, half, 4, 32);
  WSYNC();
  mm_stage(wm->sS, SS_STRIDE, 1, W + OFF_WS2, 2, 2, wm->sOut, 32, A.bs[2], false, lane);
  WSYNC();
  if (half == 0) {
    float xs[32]; float mu = 0.f;
#pragma unroll
    for (int n = 0; n < 32; ++n) {
      float r0 = valid ? A.hEs[e*SE + n] : 0.f;
      xs[n] = r0 + wm->sOut[row*32 + n];
      mu += xs[n];
    }
    mu *= (1.f/32.f);
    float var = 0.f;
#pragma unroll
    for (int n = 0; n < 32; ++n) { float d = xs[n] - mu; var += d*d; }
    var *= (1.f/32.f);
    float inv = rsqrtf(var + LNEPS);
#pragma unroll
    for (int n = 0; n < 32; ++n) {
      float o = (xs[n] - mu) * inv * A.g0[n] + A.b0[n];
      wm->sSf[row*32 + n] = o;
      wm->sS[row*SS_STRIDE + n] = (bf16)o;
    }
  }
  WSYNC();
  mm_stage(wm->sV, SV_STRIDE, 3, W + OFF_WV2, 1, 1, wm->sOut, 16, nullptr, false, lane);
  WSYNC();
  if (half == 0) {
    float xv[12]; float ss = 0.f;
#pragma unroll
    for (int o = 0; o < 4; ++o) {
      float n2 = 0.f;
#pragma unroll
      for (int t = 0; t < 3; ++t) {
        float r0 = valid ? A.hEv[e*12 + o*3 + t] : 0.f;
        float x = r0 + wm->sOut[(row*3+t)*16 + o];
        xv[o*3+t] = x; n2 += x*x;
      }
      ss += fmaxf(n2, EPSF);
    }
    float inv = rsqrtf(ss * 0.25f);
#pragma unroll
    for (int o = 0; o < 4; ++o)
#pragma unroll
      for (int t = 0; t < 3; ++t) {
        float x = xv[o*3+t] * inv;
        wm->sVf[row*12 + o*3 + t] = x;
        wm->sV[(row*3+t)*SV_STRIDE + o] = (bf16)x;
      }
  }
  WSYNC();

  // ================= f0: (32,4) -> (128,8), act =================
  mm_stage(wm->sV, SV_STRIDE, 3, W + OFF_WH3, 1, 1, wm->sOut, 48, nullptr, false, lane);
  WSYNC();
  vn_phase(wm, row, half, 8, 32);
  WSYNC();
  mm_stage(wm->sS, SS_STRIDE, 1, W + OFF_WS3, 2, 8, wm->sOut, 128, A.bs[3], true, lane);
  WSYNC();
  s_writeback(wm, row, half, 128, 160);
  WSYNC();
  mm_stage(wm->sV, SV_STRIDE, 3, W + OFF_WV3, 1, 1, wm->sOut, 16, nullptr, false, lane);
  WSYNC();
  gate_phase(wm, row, half, 8);
  WSYNC();

  // ================= f1: (128,8) -> (32,4), no act; residual + LN1 ======
  mm_stage(wm->sV, SV_STRIDE, 3, W + OFF_WH4, 1, 1, wm->sOut, 48, nullptr, false, lane);
  WSYNC();
  vn_phase(wm, row, half, 8, 128);
  WSYNC();
  mm_stage(wm->sS, SS_STRIDE, 1, W + OFF_WS4, 5, 2, wm->sOut, 32, A.bs[4], false, lane);
  WSYNC();
  if (half == 0) {
    float xs[32]; float mu = 0.f;
#pragma unroll
    for (int n = 0; n < 32; ++n) {
      xs[n] = wm->sSf[row*32 + n] + wm->sOut[row*32 + n];
      mu += xs[n];
    }
    mu *= (1.f/32.f);
    float var = 0.f;
#pragma unroll
    for (int n = 0; n < 32; ++n) { float d = xs[n] - mu; var += d*d; }
    var *= (1.f/32.f);
    float inv = rsqrtf(var + LNEPS);
    if (valid) {
#pragma unroll
      for (int n4 = 0; n4 < 8; ++n4) {
        f32x4 o4;
#pragma unroll
        for (int k = 0; k < 4; ++k) {
          int n = n4*4 + k;
          o4[k] = (xs[n] - mu) * inv * A.g1[n] + A.b1[n];
        }
        *(f32x4*)(A.outS + e*SE + n4*4) = o4;
      }
    }
  }
  WSYNC();
  mm_stage(wm->sV, SV_STRIDE, 3, W + OFF_WV4, 1, 1, wm->sOut, 16, nullptr, false, lane);
  WSYNC();
  if (half == 0) {
    float xv[12]; float ss = 0.f;
#pragma unroll
    for (int o = 0; o < 4; ++o) {
      float n2 = 0.f;
#pragma unroll
      for (int t = 0; t < 3; ++t) {
        float x = wm->sVf[row*12 + o*3 + t] + wm->sOut[(row*3+t)*16 + o];
        xv[o*3+t] = x; n2 += x*x;
      }
      ss += fmaxf(n2, EPSF);
    }
    float inv = rsqrtf(ss * 0.25f);
    if (valid) {
#pragma unroll
      for (int k4 = 0; k4 < 3; ++k4) {
        f32x4 o4;
#pragma unroll
        for (int k = 0; k < 4; ++k) o4[k] = xv[k4*4 + k] * inv;
        *(f32x4*)(A.outV + e*12 + k4*4) = o4;
      }
    }
  }
}

extern "C" void kernel_launch(void* const* d_in, const int* in_sizes, int n_in,
                              void* d_out, int out_size, void* d_ws, size_t ws_size,
                              hipStream_t stream) {
  (void)n_in; (void)out_size; (void)ws_size;
  // Detect pytree flattening order: insertion order (h_V_s first, 6.4M elems)
  // vs sorted-key order (edge_index first, 1.2M elems).
  bool sorted = (in_sizes[0] == 2 * EN);
  int iVs, iVv, iE, iEs, iEv;
  int iwh[5], iws[5], ibs[5], iwv[5], ig0, ib0, ig1, ib1;
  if (!sorted) {
    iVs = 0; iVv = 1; iE = 2; iEs = 3; iEv = 4;
    for (int L = 0; L < 5; ++L) {         // m0,m1,m2,f0,f1 ; inner wh,ws,bs,wv
      iwh[L] = 5 + 4*L; iws[L] = 6 + 4*L; ibs[L] = 7 + 4*L; iwv[L] = 8 + 4*L;
    }
    ig0 = 25; ib0 = 26; ig1 = 27; ib1 = 28;
  } else {
    iE = 0; iEs = 1; iEv = 2; iVs = 3; iVv = 4;
    // sorted: f0,f1,ln0,ln1,m0,m1,m2 ; inner bs,wh,ws,wv ; ln: b,g
    int bases[5] = {17, 21, 25, 5, 9};    // m0,m1,m2,f0,f1
    for (int L = 0; L < 5; ++L) {
      ibs[L] = bases[L] + 0; iwh[L] = bases[L] + 1;
      iws[L] = bases[L] + 2; iwv[L] = bases[L] + 3;
    }
    ib0 = 13; ig0 = 14; ib1 = 15; ig1 = 16;
  }

  unsigned short* W = (unsigned short*)d_ws;
  struct WD { int idx, N, K, Npad, Kpad, off; };
  const WD wd[15] = {
    { iwh[0], 36, 36,  48, 64,  OFF_WH0 },
    { iws[0], 32, 324, 32, 352, OFF_WS0 },
    { iwv[0], 4,  36,  16, 64,  OFF_WV0 },
    { iwh[1], 4,  4,   16, 32,  OFF_WH1 },
    { iws[1], 32, 36,  32, 64,  OFF_WS1 },
    { iwv[1], 4,  4,   16, 32,  OFF_WV1 },
    { iwh[2], 4,  4,   16, 32,  OFF_WH2 },
    { iws[2], 32, 36,  32, 64,  OFF_WS2 },
    { iwv[2], 4,  4,   16, 32,  OFF_WV2 },
    { iwh[3], 8,  4,   16, 32,  OFF_WH3 },
    { iws[3], 128,40,  128,64,  OFF_WS3 },
    { iwv[3], 8,  8,   16, 32,  OFF_WV3 },
    { iwh[4], 8,  8,   16, 32,  OFF_WH4 },
    { iws[4], 32, 136, 32, 160, OFF_WS4 },
    { iwv[4], 4,  8,   16, 32,  OFF_WV4 },
  };
  for (int t = 0; t < 15; ++t) {
    int tot = wd[t].Npad * wd[t].Kpad;
    prep_weights<<<(tot + 255) / 256, 256, 0, stream>>>(
        (const float*)d_in[wd[t].idx], W + wd[t].off,
        wd[t].N, wd[t].K, wd[t].Npad, wd[t].Kpad);
  }

  KArgs A;
  A.hVs = (const float*)d_in[iVs];
  A.hVv = (const float*)d_in[iVv];
  A.hEs = (const float*)d_in[iEs];
  A.hEv = (const float*)d_in[iEv];
  A.eidx = (const int*)d_in[iE];
  for (int L = 0; L < 5; ++L) A.bs[L] = (const float*)d_in[ibs[L]];
  A.g0 = (const float*)d_in[ig0]; A.b0 = (const float*)d_in[ib0];
  A.g1 = (const float*)d_in[ig1]; A.b1 = (const float*)d_in[ib1];
  A.W = W;
  A.outS = (float*)d_out;
  A.outV = (float*)d_out + (size_t)EN * SE;

  int grid = (EN + EPB - 1) / EPB;
  size_t shmem = sizeof(WaveMem) * WAVES;   // 235520 B (<320KB WGP LDS)
  edge_gvp_kernel<<<grid, 256, shmem, stream>>>(A);
}
